// RFA_6949257085449
// MI455X (gfx1250) — matrix operand order repaired and checked by reading, not compiled, not run on hardware
//
#include <hip/hip_runtime.h>
#include <hip/hip_bf16.h>

typedef __attribute__((ext_vector_type(16))) __bf16 v16bf;
typedef __attribute__((ext_vector_type(8)))  __bf16 v8bf;
typedef __attribute__((ext_vector_type(8)))  float  v8f;

#define BN_   4
#define NPTS  40960
#define KNBR  16
#define DIN   64
#define HCH   32
#define EPSBN 1e-5f
#define NTILE (NPTS / 16)

// Weight-tile ids in the precomputed workspace region (each tile = 32 lanes x 16 bf16 = 1KB)
#define T_W1   0   // 2 tiles
#define T_W3   2   // 4 tiles [t][c]
#define T_W4   6   // 8 tiles [t][c]
#define T_W2   14  // 4 tiles [t][c]
#define NTILES_W 18
#define WREG_BYTES (NTILES_W * 32 * 16 * 2)  // 18 KB

// ---- WMMA helper: D = A(16x32 bf16) * B(32x16 bf16) + C(16x16 f32) ----
__device__ __forceinline__ v8f wmma_bf16(v16bf a, v16bf b, v8f c) {
  return __builtin_amdgcn_wmma_f32_16x16x32_bf16(
      /*neg_a=*/false, a, /*neg_b=*/false, b,
      /*c_mod=*/(short)0, c, /*reuse_a=*/false, /*reuse_b=*/false);
}

// A-matrix (16-bit, 16x32) lane layout: lane%16 = M row.
__device__ __forceinline__ int aSlotK(int i, int h) {
  return 16 * h + i;
}

// Build one A tile of W[. x Kdim] rows [mBase..mBase+15], k-chunk kChunk,
// with BN scale g[m]*rsqrt(1+eps) folded in. Zero-pads k >= Kdim.
__device__ __forceinline__ v16bf buildAtile(const float* __restrict__ W,
                                            const float* __restrict__ g,
                                            int Kdim, int mBase, int kChunk,
                                            int lane) {
  const int h = lane >> 4;
  const int m = mBase + (lane & 15);
  const float s = g[m] * rsqrtf(1.0f + EPSBN);
  v16bf a;
#pragma unroll
  for (int i = 0; i < 16; ++i) {
    int kg = 32 * kChunk + aSlotK(i, h);
    float val = (kg < Kdim) ? W[m * Kdim + kg] * s : 0.0f;
    a[i] = (__bf16)val;
  }
  return a;
}

// ============ Stage 0: precompute all scaled A-tiles (once, 1 wave) ========
__global__ void __launch_bounds__(32) rfa_prep(
    const float* __restrict__ W1, const float* __restrict__ g1,
    const float* __restrict__ W2, const float* __restrict__ g2,
    const float* __restrict__ W3, const float* __restrict__ g3,
    const float* __restrict__ W4, const float* __restrict__ g4,
    __bf16* __restrict__ wtiles) {
  const int lane = threadIdx.x;
  v16bf* o = (v16bf*)wtiles;
#pragma unroll
  for (int t = 0; t < 2; ++t)
    o[(T_W1 + t) * 32 + lane] = buildAtile(W1, g1, 10, 16 * t, 0, lane);
#pragma unroll
  for (int t = 0; t < 2; ++t)
#pragma unroll
    for (int c = 0; c < 2; ++c)
      o[(T_W3 + 2 * t + c) * 32 + lane] = buildAtile(W3, g3, 64, 16 * t, c, lane);
#pragma unroll
  for (int t = 0; t < 4; ++t)
#pragma unroll
    for (int c = 0; c < 2; ++c)
      o[(T_W4 + 2 * t + c) * 32 + lane] = buildAtile(W4, g4, 64, 16 * t, c, lane);
#pragma unroll
  for (int t = 0; t < 2; ++t)
#pragma unroll
    for (int c = 0; c < 2; ++c)
      o[(T_W2 + 2 * t + c) * 32 + lane] = buildAtile(W2, g2, 64, 16 * t, c, lane);
}

// ================= Stage 1: f = relu(s2*(W2 @ feature) + b2) ==============
// feature [B,64,N,1] f32  ->  fpts [B][N][32] bf16 (gather-friendly layout)
__global__ void __launch_bounds__(32) rfa_stage1(
    const float* __restrict__ feature, const float* __restrict__ b2,
    const __bf16* __restrict__ wtiles, __bf16* __restrict__ fpts) {
  const int lane = threadIdx.x;
  const int h = lane >> 4, j = lane & 15;
  const int b = blockIdx.x / NTILE;
  const int nbase = (blockIdx.x % NTILE) * 16;
  const int n = nbase + j;
  const v16bf* AT = (const v16bf*)wtiles;

  v16bf A[2][2];
  float bias[2][8];
#pragma unroll
  for (int t = 0; t < 2; ++t) {
#pragma unroll
    for (int c = 0; c < 2; ++c) A[t][c] = AT[(T_W2 + 2 * t + c) * 32 + lane];
#pragma unroll
    for (int r = 0; r < 8; ++r) bias[t][r] = b2[16 * t + 8 * h + r];
  }

  // B matrix: column = point j, k = input channel (coalesced across lanes)
  v16bf Bm[2];
#pragma unroll
  for (int c = 0; c < 2; ++c) {
#pragma unroll
    for (int i = 0; i < 16; ++i) {
      int ch = 32 * c + 16 * h + i;
      Bm[c][i] = (__bf16)feature[((size_t)b * DIN + ch) * NPTS + n];
    }
  }

#pragma unroll
  for (int t = 0; t < 2; ++t) {
    v8f acc = {};
    acc = wmma_bf16(A[t][0], Bm[0], acc);
    acc = wmma_bf16(A[t][1], Bm[1], acc);
    v8bf pk;
#pragma unroll
    for (int r = 0; r < 8; ++r) pk[r] = (__bf16)fmaxf(acc[r] + bias[t][r], 0.0f);
    // rows m = 16t + 8h + r  (contiguous 8 channels -> one 16B store)
    *(v8bf*)(fpts + ((size_t)b * NPTS + n) * HCH + 16 * t + 8 * h) = pk;
  }
}

// ====== Stage 2: encoding->W1, gather, W3, maxpool-K, W4 -> out ============
__global__ void __launch_bounds__(32) rfa_stage2(
    const float* __restrict__ xyz, const int* __restrict__ nidx,
    const float* __restrict__ b1, const float* __restrict__ b3,
    const float* __restrict__ b4, const __bf16* __restrict__ wtiles,
    const __bf16* __restrict__ fpts, float* __restrict__ out) {
  __shared__ __align__(32) __bf16 fx[16][32];    // f_xyz re-layout bounce
  __shared__ __align__(32) __bf16 fcat[16][64];  // [maxpool(32) | center f(32)]

  const int lane = threadIdx.x;
  const int h = lane >> 4, j = lane & 15;
  const int b = blockIdx.x / NTILE;
  const int nbase = (blockIdx.x % NTILE) * 16;
  const v16bf* AT = (const v16bf*)wtiles;

  // ---- fetch precomputed scaled weight A-tiles (32B vector loads, L2-hot) ----
  v16bf A1[2];    float bias1[2][8];
  v16bf A3[2][2]; float bias3[2][8];
  v16bf A4[4][2]; float bias4[4][8];
#pragma unroll
  for (int t = 0; t < 2; ++t) {
    A1[t] = AT[(T_W1 + t) * 32 + lane];
#pragma unroll
    for (int c = 0; c < 2; ++c) A3[t][c] = AT[(T_W3 + 2 * t + c) * 32 + lane];
#pragma unroll
    for (int r = 0; r < 8; ++r) {
      bias1[t][r] = b1[16 * t + 8 * h + r];
      bias3[t][r] = b3[16 * t + 8 * h + r];
    }
  }
#pragma unroll
  for (int t = 0; t < 4; ++t) {
#pragma unroll
    for (int c = 0; c < 2; ++c) A4[t][c] = AT[(T_W4 + 2 * t + c) * 32 + lane];
#pragma unroll
    for (int r = 0; r < 8; ++r) bias4[t][r] = b4[16 * t + 8 * h + r];
  }

  for (int p = 0; p < 16; ++p) {
    const int n = nbase + p;
    // each lane-pair (j, j+16) owns neighbor column j
    const int idx = nidx[((size_t)b * NPTS + n) * KNBR + j];

    const float cx = xyz[((size_t)b * NPTS + n) * 3 + 0];
    const float cy = xyz[((size_t)b * NPTS + n) * 3 + 1];
    const float cz = xyz[((size_t)b * NPTS + n) * 3 + 2];
    const float nx = xyz[((size_t)b * NPTS + idx) * 3 + 0];
    const float ny = xyz[((size_t)b * NPTS + idx) * 3 + 1];
    const float nz = xyz[((size_t)b * NPTS + idx) * 3 + 2];
    const float rx = cx - nx, ry = cy - ny, rz = cz - nz;
    const float dis = sqrtf(rx * rx + ry * ry + rz * rz);
    const float enc[10] = {dis, rx, ry, rz, cx, cy, cz, nx, ny, nz};

    // B (32x16): k = 16h + i; valid k<10, rest zero-padded (matches A1 pad)
    v16bf Be;
#pragma unroll
    for (int i = 0; i < 16; ++i) {
      float v = (h == 0 && i < 10) ? enc[i] : 0.0f;
      Be[i] = (__bf16)v;
    }

    // ---- M1: f_xyz = relu(W1s @ enc + b1), stash to LDS in [nbr][ch] ----
#pragma unroll
    for (int t = 0; t < 2; ++t) {
      v8f acc = {};
      acc = wmma_bf16(A1[t], Be, acc);
      v8bf pk;
#pragma unroll
      for (int r = 0; r < 8; ++r) pk[r] = (__bf16)fmaxf(acc[r] + bias1[t][r], 0.0f);
      *(v8bf*)&fx[j][16 * t + 8 * h] = pk;  // channels 16t+8h .. +7
    }
    __syncthreads();  // single-wave block: lowers to DS waits

    // ---- B matrices for M3: gathered neighbor f (global) + f_xyz (LDS) ----
    v16bf Bn = *(const v16bf*)(fpts + ((size_t)b * NPTS + idx) * HCH + 16 * h);
    v16bf Bx = *(const v16bf*)&fx[j][16 * h];

    // ---- M3: relu(W3s @ [f_nbr; f_xyz] + b3), then max over 16 columns ----
#pragma unroll
    for (int t = 0; t < 2; ++t) {
      v8f acc = {};
      acc = wmma_bf16(A3[t][0], Bn, acc);  // k-chunk 0: neighbor features
      acc = wmma_bf16(A3[t][1], Bx, acc);  // k-chunk 1: f_xyz
#pragma unroll
      for (int r = 0; r < 8; ++r) {
        float v = fmaxf(acc[r] + bias3[t][r], 0.0f);
        v = fmaxf(v, __shfl_xor(v, 1));  // butterfly max within each 16-lane half
        v = fmaxf(v, __shfl_xor(v, 2));
        v = fmaxf(v, __shfl_xor(v, 4));
        v = fmaxf(v, __shfl_xor(v, 8));
        acc[r] = v;
      }
      if (j == 0) {  // lanes 0 & 16 publish the 16 pooled channels of this tile
        v8bf pk;
#pragma unroll
        for (int r = 0; r < 8; ++r) pk[r] = (__bf16)acc[r];
        *(v8bf*)&fcat[p][16 * t + 8 * h] = pk;
      }
    }
    // center-point features fill fcat[p][32..63]
    fcat[p][32 + lane] = fpts[((size_t)b * NPTS + n) * HCH + lane];
    __syncthreads();
  }
  __syncthreads();

  // ---- M4: out = relu(W4s @ fcat + b4) for the 16-point tile -------------
  v16bf Bc0 = *(const v16bf*)&fcat[j][16 * h];       // k = 0..31  (pooled)
  v16bf Bc1 = *(const v16bf*)&fcat[j][32 + 16 * h];  // k = 32..63 (center f)
#pragma unroll
  for (int t = 0; t < 4; ++t) {
    v8f acc = {};
    acc = wmma_bf16(A4[t][0], Bc0, acc);
    acc = wmma_bf16(A4[t][1], Bc1, acc);
#pragma unroll
    for (int r = 0; r < 8; ++r) {
      int m = 16 * t + 8 * h + r;
      out[((size_t)b * DIN + m) * NPTS + nbase + j] =
          fmaxf(acc[r] + bias4[t][r], 0.0f);
    }
  }
}

extern "C" void kernel_launch(void* const* d_in, const int* in_sizes, int n_in,
                              void* d_out, int out_size, void* d_ws, size_t ws_size,
                              hipStream_t stream) {
  const float* feature = (const float*)d_in[0];
  const float* xyz     = (const float*)d_in[1];
  const int*   nidx    = (const int*)  d_in[2];
  const float* W1 = (const float*)d_in[3];
  const float* W2 = (const float*)d_in[4];
  const float* W3 = (const float*)d_in[5];
  const float* W4 = (const float*)d_in[6];
  const float* g1 = (const float*)d_in[7];
  const float* b1 = (const float*)d_in[8];
  const float* g2 = (const float*)d_in[9];
  const float* b2 = (const float*)d_in[10];
  const float* g3 = (const float*)d_in[11];
  const float* b3 = (const float*)d_in[12];
  const float* g4 = (const float*)d_in[13];
  const float* b4 = (const float*)d_in[14];

  // workspace: [0, 32KB) scaled bf16 weight tiles; then fpts [B][N][32] bf16
  __bf16* wtiles = reinterpret_cast<__bf16*>(d_ws);
  __bf16* fpts   = reinterpret_cast<__bf16*>((char*)d_ws + 32768);
  float* out = (float*)d_out;

  dim3 grid(BN_ * NTILE), blk(32);  // one wave32 per 16-point tile
  rfa_prep<<<dim3(1), blk, 0, stream>>>(W1, g1, W2, g2, W3, g3, W4, g4, wtiles);
  rfa_stage1<<<grid, blk, 0, stream>>>(feature, b2, wtiles, fpts);
  rfa_stage2<<<grid, blk, 0, stream>>>(xyz, nidx, b1, b3, b4, wtiles, fpts, out);
}
